// GCN_Encoder1_9732395893189
// MI455X (gfx1250) — compile-verified, matrix operand order checked by reading, not run on hardware
//
#include <hip/hip_runtime.h>
#include <math.h>

// ---------------------------------------------------------------------------
// GAT encoder for MI455X (gfx1250, wave32, WMMA + Tensor Data Mover).
// GEMMs: A fp32 direct-from-global -> bf16 in regs; B pre-packed once into
// bf16 panels laid out in WMMA fragment order, then DMA'd to LDS by the TDM
// (tensor_load_to_lds), double-buffered, one barrier per K-step.
// ---------------------------------------------------------------------------

typedef __attribute__((ext_vector_type(16))) __bf16 v16bf;
typedef __attribute__((ext_vector_type(8)))  __bf16 v8bf;
typedef __attribute__((ext_vector_type(8)))  float  v8f;
typedef __attribute__((ext_vector_type(4)))  float  v4f;
typedef __attribute__((ext_vector_type(4)))  unsigned int v4u;
typedef __attribute__((ext_vector_type(8)))  int v8i;
typedef __attribute__((ext_vector_type(4)))  int v4i;

#define HID 64
#define NHEAD 2
#define BSTRIDE 40  // bf16 elements per packed-B row (80 bytes, bank-spread)

// ---- order-preserving float <-> uint key (for atomic segment-max) ----------
__device__ __forceinline__ unsigned fkey(float f) {
  unsigned u = __float_as_uint(f);
  return (u & 0x80000000u) ? ~u : (u | 0x80000000u);
}
__device__ __forceinline__ float funkey(unsigned k) {
  unsigned u = (k & 0x80000000u) ? (k & 0x7FFFFFFFu) : ~k;
  return __uint_as_float(u);
}
#define NEG_INF_KEY 0x007FFFFFu  // fkey(-inf)

// K-permutation putting a lane's 16 fragment values contiguous:
// lane(lo) wants K = {0..7, 16..23}, lane(hi) wants K = {8..15, 24..31}.
__device__ __forceinline__ int kperm(int k) {
  return (k & 7) | ((k & 8) << 1) | ((k & 16) >> 1);
}

// ---------------------------------------------------------------------------
// TDM: 1-D tile DMA global -> LDS (D# per CDNA5 ISA ch.8; groups 2/3 unused).
// ndw = tile length in DWORDs (fits tile_dim0's 16 bits for our panels).
// 6-arg builtin form (clang-23 / therock-10): extra int32x8 before cpol.
// ---------------------------------------------------------------------------
__device__ __forceinline__ void tdm_load_1d(unsigned lds_addr, const void* g,
                                            unsigned ndw) {
  unsigned long long ga = (unsigned long long)g;
  v4u g0;
  g0[0] = 1u;                                   // count=1 valid descriptor
  g0[1] = lds_addr;                             // LDS byte address
  g0[2] = (unsigned)ga;                         // global_addr[31:0]
  g0[3] = (unsigned)(ga >> 32) | (2u << 30);    // global_addr[56:32] | type=2
  v8i g1;
  g1[0] = 0x20000;                              // data_size = 4B (code 2)
  g1[1] = (int)((ndw & 0xFFFFu) << 16);         // tensor_dim0[15:0]
  g1[2] = (int)((ndw >> 16) & 0xFFFFu) | (1 << 16);  // dim0[31:16] | dim1=1
  g1[3] = (int)(ndw << 16);                     // tile_dim0 (16-bit)
  g1[4] = 0;                                    // tile_dim1/2 unused
  g1[5] = (int)ndw;                             // tensor_dim0_stride
  g1[6] = 0;
  g1[7] = 0;
  v4i z4 = {0, 0, 0, 0};
  v8i z8 = {0, 0, 0, 0, 0, 0, 0, 0};
  __builtin_amdgcn_tensor_load_to_lds(g0, g1, z4, z4, z8, 0);
}

// ---------------------------------------------------------------------------
// Pack weight matrix Bw[K x BN] (fp32) into bf16 panels, one per 32-K block,
// each panel [BN][BSTRIDE] in WMMA fragment order == the exact LDS image.
// ---------------------------------------------------------------------------
__global__ __launch_bounds__(256) void pack_b_kernel(
    const float* __restrict__ Bw, __bf16* __restrict__ out, int K, int BN) {
  int i = blockIdx.x * 256 + threadIdx.x;
  if (i >= K * BN) return;
  int k = i / BN, c = i % BN;
  int kb = k >> 5, kk = k & 31;
  out[((size_t)kb * BN + c) * BSTRIDE + kperm(kk)] = (__bf16)Bw[i];
}

// ---------------------------------------------------------------------------
// Tiled bf16 WMMA GEMM:  C[M x BN] = A[M x K] * Bpacked (+bias)
// BM=128 rows/block (8 waves x 16-row stripes), BK=32.
// B panels DMA'd by TDM, double-buffered, 1 barrier per K-step.
// ---------------------------------------------------------------------------
template <int BN, bool RELU, bool BNSTATS>
__global__ __launch_bounds__(256) void gemm_bf16_kernel(
    const float* __restrict__ A, const __bf16* __restrict__ Bp,
    const float* __restrict__ bias, float* __restrict__ C, int M, int K,
    float* __restrict__ bnsum /* [2*BN] or null */) {
  constexpr int BM = 128, BK = 32;
  constexpr int NT = BN / 16;
  constexpr int PANEL = BN * BSTRIDE;  // bf16 elems per K-block panel

  __shared__ alignas(16) __bf16 Bs[2][BN][BSTRIDE];
  __shared__ float red[BNSTATS ? 2 * BN : 1];

  const int tid  = threadIdx.x;
  const int wave = tid >> 5;
  const int lane = tid & 31;
  const int row0 = blockIdx.x * BM;

  int arow = row0 + wave * 16 + (lane & 15);
  if (arow >= M) arow = M - 1;
  const int hoff = (lane >> 4) * 8;
  const int nk = K / BK;

  v8f acc[NT];
#pragma unroll
  for (int t = 0; t < NT; ++t)
#pragma unroll
    for (int r = 0; r < 8; ++r) acc[t][r] = 0.f;

  if (BNSTATS) {
    for (int i = tid; i < 2 * BN; i += 256) red[i] = 0.f;
  }

  // prologue: DMA panel 0 into buffer 0
  if (wave == 0)
    tdm_load_1d((unsigned)(size_t)&Bs[0][0][0], Bp, PANEL / 2);

  for (int i = 0; i < nk; ++i) {
    if (wave == 0) __builtin_amdgcn_s_wait_tensorcnt(0);
    __syncthreads();  // publish buffer i&1; all waves done reading (i+1)&1
    if (wave == 0 && i + 1 < nk)
      tdm_load_1d((unsigned)(size_t)&Bs[(i + 1) & 1][0][0],
                  Bp + (size_t)(i + 1) * PANEL, PANEL / 2);

    // ---- A fragment: 4x float4 direct from global, cvt to bf16 ----
    const float* Ap = A + (size_t)arow * K + i * BK + hoff;
    v4f f0 = *(const v4f*)(Ap + 0);
    v4f f1 = *(const v4f*)(Ap + 4);
    v4f f2 = *(const v4f*)(Ap + 16);
    v4f f3 = *(const v4f*)(Ap + 20);
    if (i + 1 < nk) __builtin_prefetch(Ap + BK, 0, 1);
    v16bf a;
#pragma unroll
    for (int j = 0; j < 4; ++j) {
      a[j]      = (__bf16)f0[j];
      a[4 + j]  = (__bf16)f1[j];
      a[8 + j]  = (__bf16)f2[j];
      a[12 + j] = (__bf16)f3[j];
    }

    // ---- B fragments: hoisted contiguous LDS loads, then wmma chain ----
    v16bf bfr[NT];
#pragma unroll
    for (int t = 0; t < NT; ++t) {
      const int col = t * 16 + (lane & 15);
      v8bf lo = *(const v8bf*)&Bs[i & 1][col][2 * hoff];
      v8bf hi = *(const v8bf*)&Bs[i & 1][col][2 * hoff + 8];
#pragma unroll
      for (int j = 0; j < 8; ++j) { bfr[t][j] = lo[j]; bfr[t][8 + j] = hi[j]; }
    }
#pragma unroll
    for (int t = 0; t < NT; ++t)
      acc[t] = __builtin_amdgcn_wmma_f32_16x16x32_bf16(
          false, a, false, bfr[t], (short)0, acc[t], false, false);
  }

  // ---- epilogue: C layout VGPR r -> M=r+8*(lane>=16), lane%16 -> N ----
#pragma unroll
  for (int t = 0; t < NT; ++t) {
    const int col = t * 16 + (lane & 15);
    float csum = 0.f, csq = 0.f;
#pragma unroll
    for (int r = 0; r < 8; ++r) {
      int grow = row0 + wave * 16 + r + ((lane >> 4) * 8);
      float v = acc[t][r];
      if (bias) v += bias[col];
      if (RELU) v = v > 0.f ? v : 0.f;
      if (grow < M) {
        C[(size_t)grow * BN + col] = v;
        if (BNSTATS) { csum += v; csq += v * v; }
      }
    }
    if (BNSTATS) {
      atomicAdd(&red[col], csum);        // ds_add_f32
      atomicAdd(&red[BN + col], csq);
    }
  }
  if (BNSTATS) {
    __syncthreads();
    for (int i = tid; i < 2 * BN; i += 256) atomicAdd(&bnsum[i], red[i]);
  }
}

// ---------------------------------------------------------------------------
// vl[h] = W[:, h*64:(h+1)*64] @ attn_l[h]  (ditto vr) -> el = h . vl exactly.
// ---------------------------------------------------------------------------
__global__ __launch_bounds__(256) void prep_attn_vecs(
    const float* __restrict__ gat_w /*64x128*/, const float* __restrict__ al,
    const float* __restrict__ ar, float* __restrict__ vlr /*[4*64]*/) {
  int t = threadIdx.x;            // 256 threads
  int which = t >> 6;             // 0:vl0 1:vl1 2:vr0 3:vr1
  int k = t & 63;
  int hh = which & 1;
  const float* av = ((which >> 1) ? ar : al) + hh * HID;
  float s = 0.f;
#pragma unroll 8
  for (int d = 0; d < HID; ++d) s += gat_w[k * (NHEAD * HID) + hh * HID + d] * av[d];
  vlr[which * 64 + k] = s;
}

__global__ __launch_bounds__(256) void node_logits(
    const float* __restrict__ h, const float* __restrict__ vlr,
    float* __restrict__ el, float* __restrict__ er, int M) {
  __shared__ float v[256];
  v[threadIdx.x] = vlr[threadIdx.x];
  __syncthreads();
  int n = blockIdx.x * 256 + threadIdx.x;
  if (n >= M) return;
  float s0 = 0, s1 = 0, s2 = 0, s3 = 0;
#pragma unroll 8
  for (int k = 0; k < HID; ++k) {
    float x = h[(size_t)n * HID + k];
    s0 += x * v[k]; s1 += x * v[64 + k]; s2 += x * v[128 + k]; s3 += x * v[192 + k];
  }
  el[n * 2 + 0] = s0; el[n * 2 + 1] = s1;
  er[n * 2 + 0] = s2; er[n * 2 + 1] = s3;
}

// ---------------------------------------------------------------------------
// Per-layer scratch (re)initialization — graph-replay safe.
// ---------------------------------------------------------------------------
__global__ __launch_bounds__(256) void init_layer(
    unsigned* __restrict__ nmax, float* __restrict__ nsum,
    float* __restrict__ accum, float* __restrict__ bnsum, int n) {
  size_t i = (size_t)blockIdx.x * 256 + threadIdx.x;
  if (i < (size_t)n * 2) { nmax[i] = NEG_INF_KEY; nsum[i] = 0.f; }
  if (i < (size_t)n * 128) accum[i] = 0.f;
  if (i < 128) bnsum[i] = 0.f;
}

// ---- edge pass 1: leaky-relu logit + segment max into dst ------------------
__global__ __launch_bounds__(256) void edge_max_kernel(
    const int* __restrict__ src, const int* __restrict__ dst,
    const float* __restrict__ el, const float* __restrict__ er,
    float* __restrict__ logit, unsigned* __restrict__ nmax, int E2) {
  int i = blockIdx.x * 256 + threadIdx.x;
  if (i >= E2) return;
  int e = i >> 1, hh = i & 1;
  int s = src[e], d = dst[e];
  float v = el[s * 2 + hh] + er[d * 2 + hh];
  v = v > 0.f ? v : 0.2f * v;
  logit[i] = v;
  atomicMax(&nmax[d * 2 + hh], fkey(v));
}

// ---- edge pass 2: ex = exp(logit - max[dst]); segment sum ------------------
__global__ __launch_bounds__(256) void edge_exp_kernel(
    const int* __restrict__ dst, const unsigned* __restrict__ nmax,
    float* __restrict__ logit /* in: logit, out: ex */,
    float* __restrict__ nsum, int E2) {
  int i = blockIdx.x * 256 + threadIdx.x;
  if (i >= E2) return;
  int e = i >> 1, hh = i & 1;
  int d = dst[e];
  float ex = __expf(logit[i] - funkey(nmax[d * 2 + hh]));
  logit[i] = ex;
  atomicAdd(&nsum[d * 2 + hh], ex);
}

// ---- edge pass 3: accum[dst] += ex * feat[src] (alpha division deferred) ---
__global__ __launch_bounds__(256) void edge_scatter_kernel(
    const int* __restrict__ src, const int* __restrict__ dst,
    const float* __restrict__ ex, const float* __restrict__ feat,
    float* __restrict__ accum, long total) {
  long i = (long)blockIdx.x * 256 + threadIdx.x;
  if (i >= total) return;
  int e = (int)(i >> 7);
  int c = (int)(i & 127);
  float a = ex[e * 2 + (c >> 6)];
  atomicAdd(&accum[(size_t)dst[e] * 128 + c], a * feat[(size_t)src[e] * 128 + c]);
}

// ---- normalize by segment sum, add GAT bias, mean over heads ---------------
__global__ __launch_bounds__(256) void head_mean_kernel(
    const float* __restrict__ accum, const float* __restrict__ nsum,
    const float* __restrict__ gat_b, float* __restrict__ m, int M) {
  int i = blockIdx.x * 256 + threadIdx.x;
  if (i >= M * HID) return;
  int n = i >> 6, c = i & 63;
  float s0 = nsum[n * 2], s1 = nsum[n * 2 + 1];
  float o0 = (s0 > 0.f ? accum[(size_t)n * 128 + c] / s0 : 0.f) + gat_b[c];
  float o1 = (s1 > 0.f ? accum[(size_t)n * 128 + 64 + c] / s1 : 0.f) + gat_b[64 + c];
  m[i] = 0.5f * (o0 + o1);
}

// ---- BatchNorm normalize from fused stats ----------------------------------
__global__ __launch_bounds__(256) void bn_apply_kernel(
    const float* __restrict__ hin, const float* __restrict__ bnsum,
    const float* __restrict__ g, const float* __restrict__ b,
    float* __restrict__ out, int M) {
  int i = blockIdx.x * 256 + threadIdx.x;
  if (i >= M * HID) return;
  int c = i & 63;
  float inv = 1.f / (float)M;
  float mu = bnsum[c] * inv;
  float var = bnsum[64 + c] * inv - mu * mu;
  out[i] = (hin[i] - mu) * rsqrtf(var + 1e-5f) * g[c] + b[c];
}

// ---------------------------------------------------------------------------
extern "C" void kernel_launch(void* const* d_in, const int* in_sizes, int n_in,
                              void* d_out, int out_size, void* d_ws,
                              size_t ws_size, hipStream_t stream) {
  const float* x     = (const float*)d_in[0];
  const int*   src   = (const int*)d_in[1];
  const int*   dst   = (const int*)d_in[2];
  const float* w_dim = (const float*)d_in[3];
  const float* b_dim = (const float*)d_in[4];

  const int HIDc = in_sizes[4];            // 64
  const int IN   = in_sizes[3] / HIDc;     // 1280
  const int N    = in_sizes[0] / IN;       // 50000
  const int E    = in_sizes[1];            // 800000

  // workspace layout (floats, then bf16 panel region)
  float* ws = (float*)d_ws;
  size_t o = 0;
  float* h0    = ws + o; o += (size_t)N * 64;
  float* hl    = ws + o; o += (size_t)N * 64;
  float* feat  = ws + o; o += (size_t)N * 128;   // also reused as fc output
  float* mbuf  = ws + o; o += (size_t)N * 64;
  float* accum = ws + o; o += (size_t)N * 128;
  float* elb   = ws + o; o += (size_t)N * 2;
  float* erb   = ws + o; o += (size_t)N * 2;
  float* nsum  = ws + o; o += (size_t)N * 2;
  unsigned* nmax = (unsigned*)(ws + o); o += (size_t)N * 2;
  float* logit = ws + o; o += (size_t)E * 2;
  float* bnsum = ws + o; o += 128;
  float* vlr   = ws + o; o += 256;
  // bf16 packed-B panels (each float slot = 2 bf16)
  __bf16* pb = (__bf16*)(ws + o);
  size_t ob = 0;
  __bf16* pb_dim = pb + ob; ob += (size_t)(IN / 32) * 64 * BSTRIDE;
  __bf16* pb_gat[2]; __bf16* pb_fc[2];
  for (int l = 0; l < 2; ++l) {
    pb_gat[l] = pb + ob; ob += (size_t)2 * 128 * BSTRIDE;  // K=64 -> 2 panels
    pb_fc[l]  = pb + ob; ob += (size_t)2 * 64 * BSTRIDE;
  }
  (void)ws_size; (void)n_in; (void)out_size;

  const int gb  = (N + 127) / 128;   // GEMM row blocks
  const int nb  = (N + 255) / 256;
  const int e2b = (E * 2 + 255) / 256;

  // ---- pack all weight matrices into fragment-order bf16 panels ------------
  pack_b_kernel<<<(IN * 64 + 255) / 256, 256, 0, stream>>>(w_dim, pb_dim, IN, 64);
  for (int l = 0; l < 2; ++l) {
    pack_b_kernel<<<(64 * 128 + 255) / 256, 256, 0, stream>>>(
        (const float*)d_in[5 + l * 8 + 0], pb_gat[l], 64, 128);
    pack_b_kernel<<<(64 * 64 + 255) / 256, 256, 0, stream>>>(
        (const float*)d_in[5 + l * 8 + 4], pb_fc[l], 64, 64);
  }

  // ---- input projection: h0 = x @ w_dim + b_dim (WMMA + TDM, HBM-bound) ----
  gemm_bf16_kernel<64, false, false><<<gb, 256, 0, stream>>>(
      x, pb_dim, b_dim, h0, N, IN, nullptr);

  for (int l = 0; l < 2; ++l) {
    const float* attn_l = (const float*)d_in[5 + l * 8 + 1];
    const float* attn_r = (const float*)d_in[5 + l * 8 + 2];
    const float* gat_b  = (const float*)d_in[5 + l * 8 + 3];
    const float* fc_b   = (const float*)d_in[5 + l * 8 + 5];
    const float* bn_g   = (const float*)d_in[5 + l * 8 + 6];
    const float* bn_b   = (const float*)d_in[5 + l * 8 + 7];
    const float* gat_w  = (const float*)d_in[5 + l * 8 + 0];
    const float* hin    = (l == 0) ? h0 : hl;
    float* hout         = (l == 0) ? hl : (float*)d_out;

    init_layer<<<((size_t)N * 128 + 255) / 256, 256, 0, stream>>>(
        nmax, nsum, accum, bnsum, N);
    prep_attn_vecs<<<1, 256, 0, stream>>>(gat_w, attn_l, attn_r, vlr);
    node_logits<<<nb, 256, 0, stream>>>(hin, vlr, elb, erb, N);

    // feat = hin @ gat_w  [N x 128] (WMMA + TDM)
    gemm_bf16_kernel<128, false, false><<<gb, 256, 0, stream>>>(
        hin, pb_gat[l], nullptr, feat, N, 64, nullptr);

    edge_max_kernel<<<e2b, 256, 0, stream>>>(src, dst, elb, erb, logit, nmax, E * 2);
    edge_exp_kernel<<<e2b, 256, 0, stream>>>(dst, nmax, logit, nsum, E * 2);
    edge_scatter_kernel<<<(int)(((long)E * 128 + 255) / 256), 256, 0, stream>>>(
        src, dst, logit, feat, accum, (long)E * 128);

    head_mean_kernel<<<(N * 64 + 255) / 256, 256, 0, stream>>>(
        accum, nsum, gat_b, mbuf, N);

    // fc + ReLU + fused BN statistics (WMMA + TDM); output into feat buffer
    gemm_bf16_kernel<64, true, true><<<gb, 256, 0, stream>>>(
        mbuf, pb_fc[l], fc_b, feat, N, 64, bnsum);

    bn_apply_kernel<<<(N * 64 + 255) / 256, 256, 0, stream>>>(
        feat, bnsum, bn_g, bn_b, hout, N);
  }
}
